// ConformerBlock_86723979641003
// MI455X (gfx1250) — compile-verified, matrix operand order checked by reading, not run on hardware
//
#include <hip/hip_runtime.h>
#include <hip/hip_bf16.h>

// ---------------------------------------------------------------------------
// Conformer block forward for gfx1250 (MI455X), wave32, bf16 WMMA everywhere.
// Workspace requirement: ~98 MB (residual stream + bf16 weights + phase scratch).
// ---------------------------------------------------------------------------

typedef __attribute__((ext_vector_type(16))) __bf16 v16bf;
typedef __attribute__((ext_vector_type(8)))  float  v8f;

union FragBF { v16bf v; uint4 q[2]; };

__device__ __forceinline__ unsigned short f2bf(float f) {
    unsigned int u = __float_as_uint(f);
    u += 0x7fffu + ((u >> 16) & 1u);          // round-to-nearest-even
    return (unsigned short)(u >> 16);
}
__device__ __forceinline__ float bf2f(unsigned short h) {
    return __uint_as_float(((unsigned int)h) << 16);
}
__device__ __forceinline__ float sigmoidf_(float x) { return 1.f / (1.f + __expf(-x)); }

__device__ __forceinline__ v8f wmma_bf16(const v16bf& a, const v16bf& b, v8f c) {
    return __builtin_amdgcn_wmma_f32_16x16x32_bf16(false, a, false, b, (short)0, c, false, false);
}

#define ZERO8 {0.f,0.f,0.f,0.f,0.f,0.f,0.f,0.f}

// ---------------------------------------------------------------------------
// Weight convert: f32 [K,N] row-major -> bf16 [N,K] row-major (transposed).
// ---------------------------------------------------------------------------
__global__ void wconv_t(const float* __restrict__ in, unsigned short* __restrict__ out,
                        int K, int N) {
    int id = blockIdx.x * 256 + threadIdx.x;
    if (id >= K * N) return;
    int k = id % K, n = id / K;
    out[(size_t)n * K + k] = f2bf(in[(size_t)k * N + n]);
}

// ---------------------------------------------------------------------------
// LayerNorm over D=512: one wave per row, shfl reductions only.
// ---------------------------------------------------------------------------
template<int BF16OUT>
__global__ __launch_bounds__(256) void layernorm_k(const float* __restrict__ x,
                                                   const float* __restrict__ g,
                                                   const float* __restrict__ b,
                                                   void* __restrict__ out, int M) {
    const int lane = threadIdx.x & 31, wid = threadIdx.x >> 5;
    const int row  = blockIdx.x * 8 + wid;
    if (row >= M) return;
    const float* xr = x + (size_t)row * 512;
    float v[16];
    float s = 0.f;
#pragma unroll
    for (int i = 0; i < 16; ++i) { v[i] = xr[lane + i * 32]; s += v[i]; }
#pragma unroll
    for (int m = 16; m >= 1; m >>= 1) s += __shfl_xor(s, m, 32);
    const float mean = s * (1.f / 512.f);
    float vs = 0.f;
#pragma unroll
    for (int i = 0; i < 16; ++i) { float d = v[i] - mean; vs += d * d; }
#pragma unroll
    for (int m = 16; m >= 1; m >>= 1) vs += __shfl_xor(vs, m, 32);
    const float inv = rsqrtf(vs * (1.f / 512.f) + 1e-5f);
#pragma unroll
    for (int i = 0; i < 16; ++i) {
        int idx = lane + i * 32;
        float o = (v[i] - mean) * inv * g[idx] + b[idx];
        if (BF16OUT) ((unsigned short*)out)[(size_t)row * 512 + idx] = f2bf(o);
        else         ((float*)out)[(size_t)row * 512 + idx] = o;
    }
}

// ---------------------------------------------------------------------------
// bf16 WMMA GEMM:  out[M,N] = epilogue(A[M,K] * Bt[N,K]^T + bias)
// block = 256 thr (8 waves), tile 128x128, wave tile 32x64 (2x4 wmma tiles).
// Double-buffered LDS with register staging; one barrier per K-step.
// ---------------------------------------------------------------------------
template<int ACT_SILU_, int OUT_BF16_, int RES_>
__global__ __launch_bounds__(256)
void gemm_bf16_wmma(const unsigned short* __restrict__ A,
                    const unsigned short* __restrict__ Bt,
                    const float* __restrict__ bias,
                    const float* __restrict__ resid, float rscale,
                    void* __restrict__ out, int M, int N, int K) {
    __shared__ unsigned short As[2][128][40];   // 32 cols + 8 pad -> 80B pitch
    __shared__ unsigned short Bs[2][128][40];
    const int tid  = threadIdx.x;
    const int lane = tid & 31, wid = tid >> 5;
    const int wm   = wid & 3,  wn  = wid >> 2;
    const int half = lane >> 4, l15 = lane & 15;
    const int bm   = blockIdx.y * 128, bn = blockIdx.x * 128;
    const int ldrow = tid >> 2, ldcol = (tid & 3) << 3;   // 64 rows x 4 chunks per pass

    uint4 ra0, ra1, rb0, rb1;
    auto gload = [&](int kt) {
        ra0 = *(const uint4*)&A [(size_t)(bm +      ldrow) * K + kt + ldcol];
        ra1 = *(const uint4*)&A [(size_t)(bm + 64 + ldrow) * K + kt + ldcol];
        rb0 = *(const uint4*)&Bt[(size_t)(bn +      ldrow) * K + kt + ldcol];
        rb1 = *(const uint4*)&Bt[(size_t)(bn + 64 + ldrow) * K + kt + ldcol];
    };
    auto sstore = [&](int p) {
        *(uint4*)&As[p][     ldrow][ldcol] = ra0;
        *(uint4*)&As[p][64 + ldrow][ldcol] = ra1;
        *(uint4*)&Bs[p][     ldrow][ldcol] = rb0;
        *(uint4*)&Bs[p][64 + ldrow][ldcol] = rb1;
    };

    v8f acc[2][4];
#pragma unroll
    for (int i = 0; i < 2; ++i)
#pragma unroll
        for (int j = 0; j < 4; ++j) { v8f z = ZERO8; acc[i][j] = z; }

    gload(0); sstore(0); __syncthreads();
    int p = 0;
    for (int kt = 0; kt < K; kt += 32) {
        const bool more = (kt + 32) < K;
        if (more) gload(kt + 32);

        FragBF af[2], bfr[4];
#pragma unroll
        for (int i = 0; i < 2; ++i) {
            const unsigned short* pp = &As[p][wm * 32 + i * 16 + l15][half * 8];
            af[i].q[0] = *(const uint4*)pp;
            af[i].q[1] = *(const uint4*)(pp + 16);
        }
#pragma unroll
        for (int j = 0; j < 4; ++j) {
            const unsigned short* pp = &Bs[p][wn * 64 + j * 16 + l15][half * 8];
            bfr[j].q[0] = *(const uint4*)pp;
            bfr[j].q[1] = *(const uint4*)(pp + 16);
        }
#pragma unroll
        for (int i = 0; i < 2; ++i)
#pragma unroll
            for (int j = 0; j < 4; ++j)
                acc[i][j] = wmma_bf16(af[i].v, bfr[j].v, acc[i][j]);

        if (more) sstore(p ^ 1);
        __syncthreads();
        p ^= 1;
    }

#pragma unroll
    for (int i = 0; i < 2; ++i) {
        const int m0 = bm + wm * 32 + i * 16 + half * 8;
#pragma unroll
        for (int j = 0; j < 4; ++j) {
            const int col = bn + wn * 64 + j * 16 + l15;
            const float bv = bias[col];
#pragma unroll
            for (int r = 0; r < 8; ++r) {
                const int row = m0 + r;
                float v = acc[i][j][r] + bv;
                if (ACT_SILU_) v = v * sigmoidf_(v);
                if (RES_)      v = resid[(size_t)row * N + col] + rscale * v;
                if (OUT_BF16_) ((unsigned short*)out)[(size_t)row * N + col] = f2bf(v);
                else           ((float*)out)[(size_t)row * N + col] = v;
            }
        }
    }
}

// ---------------------------------------------------------------------------
// V transpose: v bf16 [B*S, H*DH] -> vt bf16 [B*H, DH, S]
// ---------------------------------------------------------------------------
__global__ void vt_k(const unsigned short* __restrict__ v, unsigned short* __restrict__ vt) {
    int id = blockIdx.x * 256 + threadIdx.x;       // 64*64*1024 elements
    if (id >= 64 * 64 * 1024) return;
    int s  = id & 1023;
    int d  = (id >> 10) & 63;
    int bh = id >> 16;
    int b_ = bh >> 3, h_ = bh & 7;
    vt[id] = v[(((size_t)(b_ << 10) + s) << 9) + (h_ << 6) + d];
}

// ---------------------------------------------------------------------------
// Flash attention (transposed): one wave = 16 query rows of one (b,h).
//   S^T = K_tile * Q^T  -> query lives on lanes (query = lane&15): the
//   online-softmax max/sum need only ONE __shfl_xor(16) each, and corr /
//   1/l stay lane-local for the O^T accumulation (O^T = V^T_tile * P^T).
// K fragments are software-pipelined one chunk ahead (two static buffers),
// and V fragments are issued right after the score WMMAs so both overlap
// the softmax VALU/transcendental work.
// grid = (S/128, B*H), block = 256.
// ---------------------------------------------------------------------------
__global__ __launch_bounds__(256)
void flash_attn(const unsigned short* __restrict__ q,
                const unsigned short* __restrict__ k,
                const unsigned short* __restrict__ vt,
                unsigned short* __restrict__ ao) {
    __shared__ unsigned short Pld[8][16][40];      // per-wave P scratch (16q x 32k)
    const int lane = threadIdx.x & 31, wid = threadIdx.x >> 5;
    const int half = lane >> 4, l15 = lane & 15;
    const int bh = blockIdx.y, b_ = bh >> 3, h_ = bh & 7;
    const int q0 = (blockIdx.x * 8 + wid) * 16;

    // Q fragments (lane = query row)
    FragBF qf[2];
    const size_t qrb = ((size_t)(b_ * 1024 + q0 + l15)) * 512 + h_ * 64;
#pragma unroll
    for (int kf = 0; kf < 2; ++kf) {
        qf[kf].q[0] = *(const uint4*)&q[qrb + kf * 32 + half * 8];
        qf[kf].q[1] = *(const uint4*)&q[qrb + kf * 32 + half * 8 + 16];
    }

    v8f o[4];                                      // O^T: dim = c4*16+half*8+r, query = l15
#pragma unroll
    for (int c4 = 0; c4 < 4; ++c4) { v8f z = ZERO8; o[c4] = z; }
    float mrun = -1e30f, lrun = 0.f;               // per-lane stats for query l15

    const size_t krow = ((size_t)(b_ * 1024 + l15)) * 512 + h_ * 64 + half * 8;
    const unsigned short* vbase = &vt[((size_t)(bh * 64 + l15)) * 1024 + half * 8];

    auto loadK = [&](FragBF (&kf)[2][2], int j0) {
#pragma unroll
        for (int t = 0; t < 2; ++t)
#pragma unroll
            for (int f = 0; f < 2; ++f) {
                const unsigned short* p = &k[krow + (size_t)(j0 + t * 16) * 512 + f * 32];
                kf[t][f].q[0] = *(const uint4*)p;
                kf[t][f].q[1] = *(const uint4*)(p + 16);
            }
    };

    auto process = [&](int j0, FragBF (&kf)[2][2]) {
        // S^T tiles: M = key, N = query
        v8f s0 = ZERO8, s1 = ZERO8;
        s0 = wmma_bf16(kf[0][0].v, qf[0].v, s0);
        s0 = wmma_bf16(kf[0][1].v, qf[1].v, s0);
        s1 = wmma_bf16(kf[1][0].v, qf[0].v, s1);
        s1 = wmma_bf16(kf[1][1].v, qf[1].v, s1);

        // issue V loads now: they retire during the softmax VALU/exp work
        FragBF vf[4];
#pragma unroll
        for (int c4 = 0; c4 < 4; ++c4) {
            const unsigned short* vb = vbase + (size_t)(c4 * 16) * 1024 + j0;
            vf[c4].q[0] = *(const uint4*)vb;
            vf[c4].q[1] = *(const uint4*)(vb + 16);
        }

        float a0[8], a1[8];
        float mx = -1e30f;
#pragma unroll
        for (int r = 0; r < 8; ++r) {
            a0[r] = s0[r] * 0.125f;                // scale = DH^-0.5
            a1[r] = s1[r] * 0.125f;
            mx = fmaxf(mx, fmaxf(a0[r], a1[r]));
        }
        mx = fmaxf(mx, __shfl_xor(mx, 16, 32));    // combine lane pair (other 16 keys)
        const float mnew = fmaxf(mrun, mx);
        const float corr = __expf(mrun - mnew);

        float rs = 0.f;
#pragma unroll
        for (int r = 0; r < 8; r += 2) {           // packed b32 stores of P (bf16 pairs)
            float p00 = __expf(a0[r] - mnew), p01 = __expf(a0[r + 1] - mnew);
            float p10 = __expf(a1[r] - mnew), p11 = __expf(a1[r + 1] - mnew);
            rs += (p00 + p01) + (p10 + p11);
            *(unsigned int*)&Pld[wid][l15][half * 8 + r] =
                (unsigned int)f2bf(p00) | ((unsigned int)f2bf(p01) << 16);
            *(unsigned int*)&Pld[wid][l15][16 + half * 8 + r] =
                (unsigned int)f2bf(p10) | ((unsigned int)f2bf(p11) << 16);
        }
        rs += __shfl_xor(rs, 16, 32);
        lrun = lrun * corr + rs;
        mrun = mnew;
#pragma unroll
        for (int c4 = 0; c4 < 4; ++c4)
#pragma unroll
            for (int r = 0; r < 8; ++r) o[c4][r] *= corr;

        // P as operand (row-major [16q][32k], lane = query row)
        FragBF pa;
        const unsigned short* pp = &Pld[wid][l15][half * 8];
        pa.q[0] = *(const uint4*)pp;
        pa.q[1] = *(const uint4*)(pp + 16);

        // O^T += V^T_chunk * P^T
#pragma unroll
        for (int c4 = 0; c4 < 4; ++c4)
            o[c4] = wmma_bf16(vf[c4].v, pa.v, o[c4]);
    };

    FragBF kb0[2][2], kb1[2][2];
    loadK(kb0, 0);
    for (int j0 = 0; j0 < 1024; j0 += 64) {
        loadK(kb1, j0 + 32);                       // prefetch chunk j0+32
        process(j0, kb0);
        if (j0 + 64 < 1024) loadK(kb0, j0 + 64);   // prefetch chunk j0+64
        process(j0 + 32, kb1);
    }

    const float linv = 1.f / lrun;
#pragma unroll
    for (int c4 = 0; c4 < 4; ++c4) {               // pack 8 bf16 -> one b128 store
        uint4 pkt;
        unsigned int w[4];
#pragma unroll
        for (int r2 = 0; r2 < 4; ++r2)
            w[r2] = (unsigned int)f2bf(o[c4][2 * r2] * linv) |
                    ((unsigned int)f2bf(o[c4][2 * r2 + 1] * linv) << 16);
        pkt.x = w[0]; pkt.y = w[1]; pkt.z = w[2]; pkt.w = w[3];
        const size_t base = ((size_t)(b_ * 1024 + q0 + l15)) * 512
                          + h_ * 64 + c4 * 16 + half * 8;
        *(uint4*)&ao[base] = pkt;
    }
}

// ---------------------------------------------------------------------------
// GLU: in bf16 [M,2048] -> out bf16 [M,1024]:  a * sigmoid(g)
// ---------------------------------------------------------------------------
__global__ void glu_k(const unsigned short* __restrict__ in, unsigned short* __restrict__ out,
                      int total) {
    int id = blockIdx.x * 256 + threadIdx.x;
    if (id >= total) return;
    int c = id & 1023;
    size_t row = (size_t)(id >> 10);
    float a = bf2f(in[row * 2048 + c]);
    float g = bf2f(in[row * 2048 + 1024 + c]);
    out[id] = f2bf(a * sigmoidf_(g));
}

// ---------------------------------------------------------------------------
// Depthwise conv (K=31, same pad) + BatchNorm(eval) + SiLU.
// h bf16 [B,S,CI] channel-last -> out bf16 [B,S,CI]. L2-resident working set.
// ---------------------------------------------------------------------------
__global__ void dwconv_bn_silu(const unsigned short* __restrict__ h,
                               const float* __restrict__ w,  const float* __restrict__ wb,
                               const float* __restrict__ bng, const float* __restrict__ bnb,
                               const float* __restrict__ bnm, const float* __restrict__ bnv,
                               unsigned short* __restrict__ out, int total) {
    int id = blockIdx.x * 256 + threadIdx.x;
    if (id >= total) return;
    const int c  = id & 1023;
    const int m  = id >> 10;
    const int s  = m & 1023;
    const int b_ = m >> 10;
    if (s + 16 < 1024)   // warm next rows (global_prefetch_b8)
        __builtin_prefetch(&h[(((size_t)(b_ << 10) + s + 16) << 10) + c], 0, 1);
    const float* wc = w + c * 31;
    float acc = wb[c];
#pragma unroll
    for (int kk = 0; kk < 31; ++kk) {
        int ss = s + kk - 15;
        if (ss >= 0 && ss < 1024)
            acc += bf2f(h[(((size_t)(b_ << 10) + ss) << 10) + c]) * wc[kk];
    }
    const float inv = bng[c] * rsqrtf(bnv[c] + 1e-5f);
    float y = (acc - bnm[c]) * inv + bnb[c];
    y = y * sigmoidf_(y);
    out[id] = f2bf(y);
}

// ---------------------------------------------------------------------------
// Host-side launcher
// ---------------------------------------------------------------------------
extern "C" void kernel_launch(void* const* d_in, const int* in_sizes, int n_in,
                              void* d_out, int out_size, void* d_ws, size_t ws_size,
                              hipStream_t stream) {
    (void)in_sizes; (void)n_in; (void)out_size; (void)ws_size;
    const int M = 8 * 1024;                      // B*S = 8192 rows
    const size_t MB = 1024ull * 1024ull;

    // --- inputs (setup_inputs dict order) ---
    const float* in_x     = (const float*)d_in[0];
    const float* ff1_ln_g = (const float*)d_in[1];
    const float* ff1_ln_b = (const float*)d_in[2];
    const float* ff1_w1   = (const float*)d_in[3];
    const float* ff1_b1   = (const float*)d_in[4];
    const float* ff1_w2   = (const float*)d_in[5];
    const float* ff1_b2   = (const float*)d_in[6];
    const float* attn_ln_g= (const float*)d_in[7];
    const float* attn_ln_b= (const float*)d_in[8];
    const float* q_w      = (const float*)d_in[9];
    const float* q_b      = (const float*)d_in[10];
    const float* k_w      = (const float*)d_in[11];
    const float* k_b      = (const float*)d_in[12];
    const float* v_w      = (const float*)d_in[13];
    const float* v_b      = (const float*)d_in[14];
    const float* o_w      = (const float*)d_in[15];
    const float* o_b      = (const float*)d_in[16];
    const float* conv_ln_g= (const float*)d_in[17];
    const float* conv_ln_b= (const float*)d_in[18];
    const float* pw1_w    = (const float*)d_in[19];
    const float* pw1_b    = (const float*)d_in[20];
    const float* dw_w     = (const float*)d_in[21];
    const float* dw_b     = (const float*)d_in[22];
    const float* bn_g     = (const float*)d_in[23];
    const float* bn_b     = (const float*)d_in[24];
    const float* bn_mean  = (const float*)d_in[25];
    const float* bn_var   = (const float*)d_in[26];
    const float* pw2_w    = (const float*)d_in[27];
    const float* pw2_b    = (const float*)d_in[28];
    const float* ff2_ln_g = (const float*)d_in[29];
    const float* ff2_ln_b = (const float*)d_in[30];
    const float* ff2_w1   = (const float*)d_in[31];
    const float* ff2_b1   = (const float*)d_in[32];
    const float* ff2_w2   = (const float*)d_in[33];
    const float* ff2_b2   = (const float*)d_in[34];
    const float* out_ln_g = (const float*)d_in[35];
    const float* out_ln_b = (const float*)d_in[36];

    // --- workspace arena (~98 MB peak) ---
    char* ws = (char*)d_ws;
    float*          x  = (float*)(ws);                    // f32  [M,512]  16 MB (residual)
    unsigned short* xn = (unsigned short*)(ws + 16 * MB); // bf16 [M,512]   8 MB (LN out)
    unsigned short* wt = (unsigned short*)(ws + 24 * MB); // bf16 weights  ~9 MB
    char*           sc = ws + 34 * MB;                    // phase-overlaid scratch
    // FF phases:
    unsigned short* ffh  = (unsigned short*)sc;           // bf16 [M,1024] 16 MB
    // Attention phase:
    unsigned short* qb   = (unsigned short*)sc;           // 8 MB each
    unsigned short* kb   = (unsigned short*)(sc + 8  * MB);
    unsigned short* vb   = (unsigned short*)(sc + 16 * MB);
    unsigned short* vtb  = (unsigned short*)(sc + 24 * MB);
    unsigned short* aob  = (unsigned short*)(sc + 32 * MB);
    // Conv phase:
    unsigned short* pw1o = (unsigned short*)sc;             // bf16 [M,2048] 32 MB
    unsigned short* glu  = (unsigned short*)(sc + 32 * MB); // bf16 [M,1024] 16 MB
    unsigned short* cva  = (unsigned short*)(sc + 48 * MB); // bf16 [M,1024] 16 MB

    // bf16 transposed weight slots
    size_t o = 0;
    unsigned short* wt_ff1w1 = wt + o; o += 1024 * 512;
    unsigned short* wt_ff1w2 = wt + o; o += 512 * 1024;
    unsigned short* wt_qw    = wt + o; o += 512 * 512;
    unsigned short* wt_kw    = wt + o; o += 512 * 512;
    unsigned short* wt_vw    = wt + o; o += 512 * 512;
    unsigned short* wt_ow    = wt + o; o += 512 * 512;
    unsigned short* wt_pw1   = wt + o; o += 2048 * 512;
    unsigned short* wt_pw2   = wt + o; o += 512 * 1024;
    unsigned short* wt_ff2w1 = wt + o; o += 1024 * 512;
    unsigned short* wt_ff2w2 = wt + o; o += 512 * 1024;

    auto WT = [&](const float* src, unsigned short* dst, int K, int N) {
        wconv_t<<<(K * N + 255) / 256, 256, 0, stream>>>(src, dst, K, N);
    };
    WT(ff1_w1, wt_ff1w1, 512, 1024);
    WT(ff1_w2, wt_ff1w2, 1024, 512);
    WT(q_w,    wt_qw,    512, 512);
    WT(k_w,    wt_kw,    512, 512);
    WT(v_w,    wt_vw,    512, 512);
    WT(o_w,    wt_ow,    512, 512);
    WT(pw1_w,  wt_pw1,   512, 2048);
    WT(pw2_w,  wt_pw2,   1024, 512);
    WT(ff2_w1, wt_ff2w1, 512, 1024);
    WT(ff2_w2, wt_ff2w2, 1024, 512);

    // residual stream init
    hipMemcpyAsync(x, in_x, (size_t)M * 512 * sizeof(float), hipMemcpyDeviceToDevice, stream);

    const dim3 blk(256);
    auto gN = [&](int N) { return dim3(N / 128, M / 128); };

    // ---------------- FF1 (0.5 residual) ----------------
    layernorm_k<1><<<M / 8, blk, 0, stream>>>(x, ff1_ln_g, ff1_ln_b, xn, M);
    gemm_bf16_wmma<1, 1, 0><<<gN(1024), blk, 0, stream>>>(xn, wt_ff1w1, ff1_b1, nullptr, 0.f, ffh, M, 1024, 512);
    gemm_bf16_wmma<0, 0, 1><<<gN(512),  blk, 0, stream>>>(ffh, wt_ff1w2, ff1_b2, x, 0.5f, x, M, 512, 1024);

    // ---------------- Attention ----------------
    layernorm_k<1><<<M / 8, blk, 0, stream>>>(x, attn_ln_g, attn_ln_b, xn, M);
    gemm_bf16_wmma<0, 1, 0><<<gN(512), blk, 0, stream>>>(xn, wt_qw, q_b, nullptr, 0.f, qb, M, 512, 512);
    gemm_bf16_wmma<0, 1, 0><<<gN(512), blk, 0, stream>>>(xn, wt_kw, k_b, nullptr, 0.f, kb, M, 512, 512);
    gemm_bf16_wmma<0, 1, 0><<<gN(512), blk, 0, stream>>>(xn, wt_vw, v_b, nullptr, 0.f, vb, M, 512, 512);
    vt_k<<<(64 * 64 * 1024 + 255) / 256, blk, 0, stream>>>(vb, vtb);
    flash_attn<<<dim3(8, 64), blk, 0, stream>>>(qb, kb, vtb, aob);
    gemm_bf16_wmma<0, 0, 1><<<gN(512), blk, 0, stream>>>(aob, wt_ow, o_b, x, 1.f, x, M, 512, 512);

    // ---------------- Conv module ----------------
    layernorm_k<1><<<M / 8, blk, 0, stream>>>(x, conv_ln_g, conv_ln_b, xn, M);
    gemm_bf16_wmma<0, 1, 0><<<gN(2048), blk, 0, stream>>>(xn, wt_pw1, pw1_b, nullptr, 0.f, pw1o, M, 2048, 512);
    glu_k<<<(M * 1024 + 255) / 256, blk, 0, stream>>>(pw1o, glu, M * 1024);
    dwconv_bn_silu<<<(M * 1024 + 255) / 256, blk, 0, stream>>>(glu, dw_w, dw_b, bn_g, bn_b, bn_mean, bn_var, cva, M * 1024);
    gemm_bf16_wmma<0, 0, 1><<<gN(512), blk, 0, stream>>>(cva, wt_pw2, pw2_b, x, 1.f, x, M, 512, 1024);

    // ---------------- FF2 (0.5 residual) ----------------
    layernorm_k<1><<<M / 8, blk, 0, stream>>>(x, ff2_ln_g, ff2_ln_b, xn, M);
    gemm_bf16_wmma<1, 1, 0><<<gN(1024), blk, 0, stream>>>(xn, wt_ff2w1, ff2_b1, nullptr, 0.f, ffh, M, 1024, 512);
    gemm_bf16_wmma<0, 0, 1><<<gN(512),  blk, 0, stream>>>(ffh, wt_ff2w2, ff2_b2, x, 0.5f, x, M, 512, 1024);

    // ---------------- Final LayerNorm -> d_out (f32) ----------------
    layernorm_k<0><<<M / 8, blk, 0, stream>>>(x, out_ln_g, out_ln_b, d_out, M);
}
